// FastEuclideanAttention_51539607552999
// MI455X (gfx1250) — matrix operand order repaired and checked by reading, not compiled, not run on hardware
//
#include <hip/hip_runtime.h>

#define BB 4
#define LL 1024
#define HH 12
#define EE 64
#define NT (LL / 16)

typedef __attribute__((ext_vector_type(16))) _Float16 v16h;
typedef __attribute__((ext_vector_type(8)))  float    v8f;
typedef __attribute__((ext_vector_type(4)))  float    v4f;

// ---------------------------------------------------------------------------
// Kernel 1: s[b,h,j] = qscale * dot(queries[b,j,h,:], w[h,:])
// One wave32 per (b,h,j); lane t owns elements t and t+32.
// ---------------------------------------------------------------------------
__global__ __launch_bounds__(256)
void fea_scores_kernel(const float* __restrict__ q, const float* __restrict__ qmask,
                       const float* __restrict__ w, float* __restrict__ s_arr) {
  int wid  = (int)((blockIdx.x * 256u + threadIdx.x) >> 5);
  int lane = threadIdx.x & 31;
  int j  = wid & (LL - 1);
  int bh = wid >> 10;                    // LL == 1024
  int h = bh % HH, b = bh / HH;
  const float* qp = q + (((size_t)b * LL + j) * HH + h) * EE;
  const float* wp = w + (size_t)h * EE;
  float part = qp[lane] * wp[lane] + qp[lane + 32] * wp[lane + 32];
#pragma unroll
  for (int d = 16; d >= 1; d >>= 1) part += __shfl_xor(part, d, 32);
  if (lane == 0) s_arr[(size_t)bh * LL + j] = part * qmask[(size_t)b * LL];
}

// ---------------------------------------------------------------------------
// Kernel 2: prefix softmax scan over j for each (b,h):
//   m_i = max_{j<=i} s_j ,  rz_i = 1 / sum_{j<=i} exp(s_j - m_i)
// Lane t owns 32 consecutive j; local scan + log2 wave combine + re-walk.
// ---------------------------------------------------------------------------
__global__ __launch_bounds__(32)
void fea_scan_kernel(const float* __restrict__ s_arr,
                     float* __restrict__ m_arr, float* __restrict__ rz_arr) {
  const float NEG = -1e30f;
  int bh = blockIdx.x;
  int t  = threadIdx.x;
  const float* s = s_arr + (size_t)bh * LL;
  int j0 = t * 32;
  float m = NEG, z = 0.0f;
  for (int kx = 0; kx < 32; ++kx) {
    float sj = s[j0 + kx];
    float mn = fmaxf(m, sj);
    z = z * __expf(m - mn) + __expf(sj - mn);
    m = mn;
  }
  // inclusive wave scan of (m,z) aggregates
  float M = m, Z = z;
#pragma unroll
  for (int d = 1; d < 32; d <<= 1) {
    float Mo = __shfl_up(M, d, 32);
    float Zo = __shfl_up(Z, d, 32);
    if (t >= d) {
      float Mn = fmaxf(M, Mo);
      Z = Z * __expf(M - Mn) + Zo * __expf(Mo - Mn);
      M = Mn;
    }
  }
  // exclusive prefix for this lane
  float Me = __shfl_up(M, 1, 32);
  float Ze = __shfl_up(Z, 1, 32);
  if (t == 0) { Me = NEG; Ze = 0.0f; }
  m = Me; z = Ze;
  for (int kx = 0; kx < 32; ++kx) {
    float sj = s[j0 + kx];
    float mn = fmaxf(m, sj);
    z = z * __expf(m - mn) + __expf(sj - mn);
    m = mn;
    m_arr [(size_t)bh * LL + j0 + kx] = m;
    rz_arr[(size_t)bh * LL + j0 + kx] = 1.0f / z;
  }
}

// ---------------------------------------------------------------------------
// Kernel 3: one wave32 per (b, h, 16-query tile).
//  Phase A: qhat tile = P1 x Q  (WMMA), staged via LDS into B-fragments.
//  Phase B: causal flash attention, scores S^T = K x Qhat^T (WMMA),
//           online softmax, O += P x V (WMMA). out = queries + O / l.
// ---------------------------------------------------------------------------
__global__ __launch_bounds__(32)
void fea_attn_kernel(const float* __restrict__ q, const float* __restrict__ kk,
                     const float* __restrict__ vv, const float* __restrict__ kmask,
                     const float* __restrict__ s_arr, const float* __restrict__ m_arr,
                     const float* __restrict__ rz_arr, float* __restrict__ out) {
  __shared__ __align__(16) float lq[16 * EE];       // qhat tile staging (4 KB)
  const float NEG = -1e30f;
  const int gid = blockIdx.x;
  const int it  = gid % NT;
  const int bh  = gid / NT;
  const int h = bh % HH, b = bh / HH;
  const int lane = threadIdx.x;
  const int n = lane & 15;                          // column within 16
  const int g = lane >> 4;                          // half-wave
  const int i0 = it * 16;
  const int irow = i0 + n;                          // this lane's i (stats/columns)

  const size_t rowstride = (size_t)HH * EE;
  const float* qbase = q  + ((size_t)b * LL * HH + h) * EE;
  const float* kbase = kk + ((size_t)b * LL * HH + h) * EE;
  const float* vbase = vv + ((size_t)b * LL * HH + h) * EE;
  float*       obase = out + ((size_t)b * LL * HH + h) * EE;

  const float* sBH  = s_arr + (size_t)bh * LL;
  const float  m_i  = m_arr [(size_t)bh * LL + irow];
  const float  rz_i = rz_arr[(size_t)bh * LL + irow];
  const float  kscale = kmask[(size_t)b * LL];

  const int njb = ((i0 + 15) >> 5) + 1;             // 32-j blocks covering j <= i0+15

  // ---------------- Phase A: build qhat tile ----------------
  v8f accA[4] = {};
  for (int jb = 0; jb < njb; ++jb) {
    const int j0 = jb * 32;
    // A-fragment: P1[i][j] = exp(s_j - m_i) * rz_i  (causal-masked).
    // Loads are unconditionally in-bounds (j <= LL-1): load first, select after
    // so the compiler batches 16 plain loads instead of predicated branches.
    float sv[16];
#pragma unroll
    for (int e2 = 0; e2 < 16; ++e2) {
      int kidx = ((e2 >> 3) << 4) + (e2 & 7) + 8 * g;     // A 16-bit layout
      sv[e2] = sBH[j0 + kidx];
    }
    v16h aP;
#pragma unroll
    for (int e2 = 0; e2 < 16; ++e2) {
      int kidx = ((e2 >> 3) << 4) + (e2 & 7) + 8 * g;
      int j = j0 + kidx;
      float p = __expf(sv[e2] - m_i) * rz_i;
      aP[e2] = (_Float16)((j <= irow) ? p : 0.0f);
    }
#pragma unroll
    for (int c = 0; c < 4; ++c) {
      // B-fragment: Q chunk, lane holds column e = c*16+n, K rows j = 16*(e2>>3) + (e2&7) + 8g
      v16h bQ;
#pragma unroll
      for (int e2 = 0; e2 < 16; ++e2) {
        int j = j0 + ((e2 >> 3) << 4) + (e2 & 7) + 8 * g;
        bQ[e2] = (_Float16)qbase[(size_t)j * rowstride + c * 16 + n];
      }
      accA[c] = __builtin_amdgcn_wmma_f32_16x16x32_f16(
          false, aP, false, bQ, (short)0, accA[c], false, false);
    }
  }
  // stage qhat (D layout: lane -> e = c*16+n, rows i = r+8g) into LDS
#pragma unroll
  for (int c = 0; c < 4; ++c)
#pragma unroll
    for (int r = 0; r < 8; ++r)
      lq[(r + 8 * g) * EE + c * 16 + n] = accA[c][r];
  __syncthreads();
  // reload as B-fragments of Qhat^T: lane holds column i = n, K(e) rows = f*32+8g+16*(e2>>3)+(e2&7)
  v16h bQh[2];
  {
    const v4f* lq4 = (const v4f*)lq;
#pragma unroll
    for (int f = 0; f < 2; ++f)
#pragma unroll
      for (int u = 0; u < 4; ++u) {
        v4f x = lq4[(n * EE + f * 32 + 8 * g + ((u >> 1) << 4)) / 4 + (u & 1)];
#pragma unroll
        for (int c4 = 0; c4 < 4; ++c4) bQh[f][u * 4 + c4] = (_Float16)x[c4];
      }
  }

  // ---------------- Phase B: causal euclidean flash attention ----------------
  v8f accO[4] = {};
  float mrun = NEG, lrun = 0.0f;

  for (int jb = 0; jb < njb; ++jb) {
    const int j0 = jb * 32;
    v8f sc[2];
#pragma unroll
    for (int t = 0; t < 2; ++t) {
      const int jrow = j0 + t * 16 + n;             // K row held by this lane
      const v4f* kp4 = (const v4f*)(kbase + (size_t)jrow * rowstride);
      v16h aK[2];
      float knp = 0.0f;
#pragma unroll
      for (int f = 0; f < 2; ++f) {                 // A layout: runs of 8 -> 2x b128
        v4f a0 = kp4[f * 8 + 2 * g];
        v4f a1 = kp4[f * 8 + 2 * g + 1];
        v4f b0 = kp4[f * 8 + 4 + 2 * g];
        v4f b1 = kp4[f * 8 + 4 + 2 * g + 1];
#pragma unroll
        for (int u = 0; u < 4; ++u) {
          knp += a0[u] * a0[u] + a1[u] * a1[u] + b0[u] * b0[u] + b1[u] * b1[u];
          aK[f][u]      = (_Float16)a0[u];
          aK[f][u + 4]  = (_Float16)a1[u];
          aK[f][u + 8]  = (_Float16)b0[u];
          aK[f][u + 12] = (_Float16)b1[u];
        }
      }
      float kn = knp + __shfl_xor(knp, 16, 32);     // ||k_j||^2 for local row n
      // S^T subtile = K(16x64) x Qhat^T(64x16), f32 accum
      v8f s0 = {};
      s0 = __builtin_amdgcn_wmma_f32_16x16x32_f16(false, aK[0], false, bQh[0],
                                                  (short)0, s0, false, false);
      s0 = __builtin_amdgcn_wmma_f32_16x16x32_f16(false, aK[1], false, bQh[1],
                                                  (short)0, s0, false, false);
#pragma unroll
      for (int r = 0; r < 8; ++r) {
        float knr = __shfl(kn, r + 8 * g, 32);      // kn of row j = r+8g
        float d = kscale * (2.0f * s0[r] - knr);    // qn_i term drops (row const)
        int j = j0 + t * 16 + r + 8 * g;
        sc[t][r] = (j <= irow) ? d : NEG;           // causal mask (pure select)
      }
    }
    // online softmax over this block's 32 j (per column i = irow)
    float bm = NEG;
#pragma unroll
    for (int t = 0; t < 2; ++t)
#pragma unroll
      for (int r = 0; r < 8; ++r) bm = fmaxf(bm, sc[t][r]);
    bm = fmaxf(bm, __shfl_xor(bm, 16, 32));
    const float mnew = fmaxf(mrun, bm);
    const float corr = __expf(mrun - mnew);
    mrun = mnew;
    // P: D layout of S^T lands exactly in the f16 A-fragment layout (K=32)
    v16h aP2;
    float rs = 0.0f;
#pragma unroll
    for (int t = 0; t < 2; ++t)
#pragma unroll
      for (int r = 0; r < 8; ++r) {
        float p = __expf(sc[t][r] - mnew);
        rs += p;
        aP2[t * 8 + r] = (_Float16)p;
      }
    rs += __shfl_xor(rs, 16, 32);
    lrun = lrun * corr + rs;
    float cb[8];
#pragma unroll
    for (int r = 0; r < 8; ++r) cb[r] = __shfl(corr, r + 8 * g, 32);
#pragma unroll
    for (int c = 0; c < 4; ++c)
#pragma unroll
      for (int r = 0; r < 8; ++r) accO[c][r] *= cb[r];
    // O += P x V
#pragma unroll
    for (int c = 0; c < 4; ++c) {
      v16h bV;
#pragma unroll
      for (int e2 = 0; e2 < 16; ++e2) {
        int j = j0 + ((e2 >> 3) << 4) + (e2 & 7) + 8 * g;
        bV[e2] = (_Float16)vbase[(size_t)j * rowstride + c * 16 + n];
      }
      accO[c] = __builtin_amdgcn_wmma_f32_16x16x32_f16(
          false, aP2, false, bV, (short)0, accO[c], false, false);
    }
  }

  const float inv = 1.0f / lrun;
#pragma unroll
  for (int r = 0; r < 8; ++r) {
    const float ib = __shfl(inv, r + 8 * g, 32);
    const int i = i0 + r + 8 * g;
#pragma unroll
    for (int c = 0; c < 4; ++c) {
      const size_t o = (size_t)i * rowstride + c * 16 + n;
      obase[o] = qbase[o] + accO[c][r] * ib;
    }
  }
}

// ---------------------------------------------------------------------------
extern "C" void kernel_launch(void* const* d_in, const int* in_sizes, int n_in,
                              void* d_out, int out_size, void* d_ws, size_t ws_size,
                              hipStream_t stream) {
  (void)in_sizes; (void)n_in; (void)out_size; (void)ws_size;
  const float* q  = (const float*)d_in[0];
  const float* k  = (const float*)d_in[1];
  const float* v  = (const float*)d_in[2];
  const float* qm = (const float*)d_in[3];
  const float* km = (const float*)d_in[4];
  const float* w  = (const float*)d_in[5];
  float* outp = (float*)d_out;

  float* s_arr  = (float*)d_ws;                       // [B*H*L]
  float* m_arr  = s_arr + (size_t)BB * HH * LL;       // [B*H*L]
  float* rz_arr = m_arr + (size_t)BB * HH * LL;       // [B*H*L]

  fea_scores_kernel<<<dim3((BB * HH * LL * 32) / 256), dim3(256), 0, stream>>>(
      q, qm, w, s_arr);
  fea_scan_kernel<<<dim3(BB * HH), dim3(32), 0, stream>>>(s_arr, m_arr, rz_arr);
  fea_attn_kernel<<<dim3(BB * HH * NT), dim3(32), 0, stream>>>(
      q, k, v, km, s_arr, m_arr, rz_arr, outp);
}
